// GeometricMoEFFN_8186207666927
// MI455X (gfx1250) — matrix operand order repaired and checked by reading, not compiled, not run on hardware
//
#include <hip/hip_runtime.h>

// ---------------------------------------------------------------------------
// GeometricMoEFFN for MI455X (gfx1250, wave32, WMMA, TDM).
// Heavy math: top-2 MoE grouped GEMM in bf16 WMMA (f32 accum), 6.4 GFLOP
// instead of the reference's dense 25.8 GFLOP. Stage-2 down-projection
// weights are DMA'd into LDS by the Tensor Data Mover (overlapped with the
// stage-1 gate/up WMMAs), with TDM row padding for conflict-free ds_load_b128.
// ---------------------------------------------------------------------------

typedef __attribute__((ext_vector_type(16))) __bf16 v16bf;
typedef __attribute__((ext_vector_type(8)))  __bf16 v8bf;
typedef __attribute__((ext_vector_type(8)))  float  v8f;
typedef unsigned int u32x4 __attribute__((ext_vector_type(4)));
typedef int          i32x4 __attribute__((ext_vector_type(4)));
typedef int          i32x8 __attribute__((ext_vector_type(8)));

typedef __attribute__((address_space(3))) void lds_void;
typedef __attribute__((address_space(3))) char lds_char;
typedef __attribute__((address_space(1))) void gbl_void;

#define EPS_RMS 1.1920929e-07f

static constexpr int NTOK = 4096;            // B*T
static constexpr int NB   = 8;               // blades
static constexpr int D    = 64;
static constexpr int NE   = 8;               // experts
static constexpr int FF   = 256;
static constexpr int NGRP = NB * NE;         // 64 (blade,expert) groups
static constexpr int NASSIGN = NTOK * NB * 2;  // 65536 top-2 assignments
static constexpr int WEL  = NB * NE * FF * D;  // 1,048,576 weight elems

#if __has_builtin(__builtin_amdgcn_tensor_load_to_lds)
#define HAVE_TDM 1
#else
#define HAVE_TDM 0
#endif
#if __has_builtin(__builtin_amdgcn_global_load_async_to_lds_b128)
#define HAVE_ASYNC 1
#else
#define HAVE_ASYNC 0
#endif
#define USE_LDS_DW (HAVE_TDM || HAVE_ASYNC)

__device__ inline v8f wmma_bf16(v16bf a, v16bf b, v8f c) {
  // v_wmma_f32_16x16x32_bf16: D = A(16x32) * B(32x16) + C, f32 accumulate
  return __builtin_amdgcn_wmma_f32_16x16x32_bf16(
      /*neg_a=*/false, a, /*neg_b=*/false, b,
      /*c_mod=*/(short)0, c, /*reuse_a=*/false, /*reuse_b=*/false);
}

// A-matrix register packing used by this code (16-bit A 16x32, wave32):
// lane half h = lane/16; elements a[0..15] hold K = kbase + 16h + 0..15
//               (contiguous, matching the B operand fills).
__device__ inline v16bf pack_a(const __bf16* row, int kbase, int h) {
  v8bf lo = *(const v8bf*)(row + kbase + 16 * h);
  v8bf hi = *(const v8bf*)(row + kbase + 16 * h + 8);
  v16bf r;
#pragma unroll
  for (int i = 0; i < 8; ++i) { r[i] = lo[i]; r[i + 8] = hi[i]; }
  return r;
}

// B operand from a 16-byte-aligned source (two b128 loads, concatenated Ks).
__device__ inline v16bf pack_b(const __bf16* p) {
  v8bf lo = *(const v8bf*)p;
  v8bf hi = *(const v8bf*)(p + 8);
  v16bf r;
#pragma unroll
  for (int i = 0; i < 8; ++i) { r[i] = lo[i]; r[i + 8] = hi[i]; }
  return r;
}

// ---------------------------------------------------------------------------
__global__ void k_init(float* f_sum, int* counts, int* cursor) {
  int i = threadIdx.x;
  if (i < NGRP) { f_sum[i] = 0.f; counts[i] = 0; cursor[i] = 0; }
}

__global__ void k_f32_to_bf16(const float* __restrict__ in,
                              __bf16* __restrict__ out, int n) {
  int i = blockIdx.x * blockDim.x + threadIdx.x;
  if (i < n) out[i] = (__bf16)in[i];
}

// ---------------------------------------------------------------------------
// Per-token: RMSNorm-in, Cayley geometric product, router scores, softmax,
// top-2, aux accumulators. 256 threads = 8 waves, wave b owns blade b.
__global__ void k_token_prep(const float* __restrict__ x,
                             const float* __restrict__ inter,     // (8,8)
                             const float* __restrict__ norm_in_w, // (512,)
                             const float* __restrict__ router_w,  // (8,8,64)
                             float*  __restrict__ bladesF,        // (4096,8,64)
                             __bf16* __restrict__ bladesB,        // (4096,8,64)
                             int*    __restrict__ topIdx,         // (65536,)
                             float*  __restrict__ topW,           // (65536,)
                             float*  __restrict__ f_sum,          // (64,)
                             int*    __restrict__ counts)         // (64,)
{
  const int t = blockIdx.x;
  const int tid = threadIdx.x;
  const int wave = tid >> 5, lane = tid & 31;
  const int b = wave;

  __shared__ float s_bl[512];  // normed blades, [blade*64 + d]
  __shared__ float s_sw[64];   // cayley sign * sigmoid(interaction)
  __shared__ float s_red[8];

  const float* xt = x + (size_t)t * 512;
  float v0 = xt[2 * tid], v1 = xt[2 * tid + 1];
  float ss = v0 * v0 + v1 * v1;
#pragma unroll
  for (int off = 16; off; off >>= 1) ss += __shfl_xor(ss, off, 32);
  if (lane == 0) s_red[wave] = ss;

  if (tid < 64) {
    int i = tid >> 3, j = tid & 7;
    int s = 0, aa = i >> 1;
    while (aa) { s += __popc(aa & j); aa >>= 1; }
    float sg = (s & 1) ? -1.f : 1.f;
    s_sw[tid] = sg / (1.f + expf(-inter[tid]));
  }
  __syncthreads();

  float tot = 0.f;
#pragma unroll
  for (int i = 0; i < 8; ++i) tot += s_red[i];
  float rinv = rsqrtf(tot * (1.f / 512.f) + EPS_RMS);
  s_bl[2 * tid]     = v0 * rinv * norm_in_w[2 * tid];
  s_bl[2 * tid + 1] = v1 * rinv * norm_in_w[2 * tid + 1];
  __syncthreads();

  // geometric product for output blade k=b at d0, d0+1 (2 cols per lane)
  const int d0 = lane * 2;
  float g0 = 0.f, g1 = 0.f;
#pragma unroll
  for (int i = 0; i < 8; ++i) {
    int j = i ^ b;
    float sw = s_sw[i * 8 + j];
    g0 += sw * s_bl[i * 64 + d0]     * s_bl[j * 64 + d0];
    g1 += sw * s_bl[i * 64 + d0 + 1] * s_bl[j * 64 + d0 + 1];
  }
  float p0 = s_bl[b * 64 + d0] + g0;
  float p1 = s_bl[b * 64 + d0 + 1] + g1;
  size_t obase = ((size_t)t * NB + b) * 64 + d0;
  bladesF[obase] = p0;  bladesF[obase + 1] = p1;
  bladesB[obase] = (__bf16)p0;  bladesB[obase + 1] = (__bf16)p1;

  // router scores for this blade: 8 experts, wave-butterfly reduction
  float sc[8];
  const float* rw = router_w + (size_t)b * NE * 64;
#pragma unroll
  for (int e = 0; e < 8; ++e)
    sc[e] = p0 * rw[e * 64 + d0] + p1 * rw[e * 64 + d0 + 1];
#pragma unroll
  for (int off = 16; off; off >>= 1)
#pragma unroll
    for (int e = 0; e < 8; ++e) sc[e] += __shfl_xor(sc[e], off, 32);

  // softmax (replicated on all lanes; cheap)
  float m = sc[0];
#pragma unroll
  for (int e = 1; e < 8; ++e) m = fmaxf(m, sc[e]);
  float pr[8], sum = 0.f;
#pragma unroll
  for (int e = 0; e < 8; ++e) { pr[e] = expf(sc[e] - m); sum += pr[e]; }
  float isum = 1.f / sum;
#pragma unroll
  for (int e = 0; e < 8; ++e) pr[e] *= isum;

  if (lane < 8) atomicAdd(&f_sum[b * 8 + lane], pr[lane]);  // f accumulator

  if (lane == 0) {
    int e0 = 0;
#pragma unroll
    for (int e = 1; e < 8; ++e) if (pr[e] > pr[e0]) e0 = e;
    int e1 = (e0 == 0) ? 1 : 0;
#pragma unroll
    for (int e = 0; e < 8; ++e) if (e != e0 && pr[e] > pr[e1]) e1 = e;
    float w0 = pr[e0], w1 = pr[e1];
    float r = 1.f / (w0 + w1 + 1e-8f);
    int ib = (t * NB + b) * 2;
    topIdx[ib] = e0;       topIdx[ib + 1] = e1;
    topW[ib]   = w0 * r;   topW[ib + 1]   = w1 * r;
    atomicAdd(&counts[b * 8 + e0], 1);   // doubles as "p" accumulator
    atomicAdd(&counts[b * 8 + e1], 1);
  }
}

// ---------------------------------------------------------------------------
__global__ void k_prefix(const int* __restrict__ counts, int* __restrict__ offsets) {
  if (threadIdx.x == 0) {
    int a = 0;
    for (int i = 0; i < NGRP; ++i) { offsets[i] = a; a += counts[i]; }
  }
}

__global__ void k_assign(const int* __restrict__ topIdx,
                         const float* __restrict__ topW,
                         const int* __restrict__ offsets,
                         int* __restrict__ cursor,
                         int* __restrict__ rows, float* __restrict__ wts,
                         int* __restrict__ slotOf) {
  int idx = blockIdx.x * blockDim.x + threadIdx.x;   // == (t*8+b)*2+k
  if (idx >= NASSIGN) return;
  int rem = idx & 15;
  int b = rem >> 1;
  int t = idx >> 4;
  int e = topIdx[idx];
  int g = b * 8 + e;
  int pos = atomicAdd(&cursor[g], 1);
  int slot = offsets[g] + pos;
  rows[slot] = t;
  wts[slot] = topW[idx];
  slotOf[idx] = slot;
}

// ---------------------------------------------------------------------------
// Grouped MoE GEMM. Grid: (128 row-tiles of 32 tokens, 64 groups), 8 waves.
// Prologue: TDM (or async-LDS) DMA of the 64x256 bf16 down-weight tile into
//           padded LDS, overlapped with stage 1.
// Stage 1: hidden[32,256] = silu(A*gateB)*(A*upB)  -> LDS (bf16, padded)
// Stage 2: out[32,64]     = hidden * down^T        -> expertOut slots (f32)
__global__ void k_moe_gemm(const __bf16* __restrict__ bladesB,
                           const __bf16* __restrict__ gateB,   // [g][F][D]
                           const __bf16* __restrict__ upB,     // [g][F][D]
                           const __bf16* __restrict__ downB,   // [g][D][F]
                           const int* __restrict__ rows,
                           const int* __restrict__ offsets,
                           const int* __restrict__ counts,
                           float* __restrict__ expertOut)      // [slot][64]
{
  const int g = blockIdx.y;
  const int b = g >> 3;
  const int base = offsets[g];
  const int count = counts[g];
  const int m0 = blockIdx.x * 32;
  if (m0 >= count) return;

  const int tid = threadIdx.x, wave = tid >> 5, lane = tid & 31;
  const int h = lane >> 4, ln = lane & 15;

  __shared__ int    s_tok[32];
  __shared__ __bf16 s_hid[32][FF + 8];   // +8 bf16 pad: conflict-free b128
  __shared__ __bf16 s_dw[D][FF + 8];     // down weights, same padding

#if HAVE_TDM
  // ---- prologue: one TDM descriptor moves the whole 64x256 bf16 tile.
  // D# pad fields insert 16B after every 512B row -> matches s_dw stride.
  if (wave == 0) {
    const __bf16* dwg = downB + (size_t)g * (D * FF);
    unsigned long long ga = (unsigned long long)(uintptr_t)dwg;
    unsigned lds_off =
        (unsigned)(unsigned long long)(lds_char*)&s_dw[0][0];
    u32x4 g0 = {
        1u,                                    // count=1 valid descriptor
        lds_off,                               // lds_addr (bytes)
        (unsigned)(ga & 0xffffffffu),          // global_addr[31:0]
        (unsigned)((ga >> 32) & 0x01ffffffu) | (2u << 30)  // addr[56:32]|type=2
    };
    i32x8 g1 = {
        (1 << 16) |                            // data_size = 1 (2 bytes)
        (1 << 20) |                            // pad_enable
        (6 << 22) |                            // pad_interval: 128 dwords
        (3 << 25),                             // pad_amount: 4 dwords (16B)
        (FF << 16),                            // tensor_dim0[15:0] @ [63:48]
        (D << 16),                             // tensor_dim1[15:0] @ [95:80]
        (FF << 16),                            // tile_dim0 @ [127:112]
        D,                                     // tile_dim1 @ [143:128]
        FF,                                    // tensor_dim0_stride[31:0]
        0, 0
    };
    i32x4 z4 = {0, 0, 0, 0};
#if __clang_major__ >= 23
    i32x8 z8 = {0, 0, 0, 0, 0, 0, 0, 0};
    __builtin_amdgcn_tensor_load_to_lds(g0, g1, z4, z4, z8, 0);
#else
    __builtin_amdgcn_tensor_load_to_lds(g0, g1, z4, z4, 0);
#endif
  }
#elif HAVE_ASYNC
  { // ---- prologue: per-lane async copies (ASYNCcnt), 2048 x 16B chunks.
    const __bf16* dwg = downB + (size_t)g * (D * FF);
#pragma unroll
    for (int it = 0; it < 8; ++it) {
      int idx = it * 256 + tid;
      int row = idx >> 5, c = idx & 31;
      const __bf16* src = dwg + row * FF + c * 8;
      __bf16* dst = &s_dw[row][c * 8];
      __builtin_amdgcn_global_load_async_to_lds_b128(
          (gbl_void*)(void*)src, (lds_void*)dst, 0, 0);
    }
  }
#endif

  if (tid < 32) {
    int r = m0 + tid;
    s_tok[tid] = rows[base + (r < count ? r : 0)];
  }
  __syncthreads();

  { // ---- stage 1: gate/up GEMMs, K = D = 64 (overlaps the DMA) ----
    const int Mblk = wave >> 2;          // token half: 0..1
    const int nch  = wave & 3;           // 64-wide ff chunk: 0..3
    const int tok  = s_tok[Mblk * 16 + ln];
    const __bf16* arow = bladesB + ((size_t)tok * NB + b) * 64;
    v16bf A0 = pack_a(arow, 0, h);
    v16bf A1 = pack_a(arow, 32, h);
    const __bf16* gW = gateB + (size_t)g * (FF * D);
    const __bf16* uW = upB   + (size_t)g * (FF * D);
#pragma unroll
    for (int nt = 0; nt < 4; ++nt) {
      const int f = nch * 64 + nt * 16 + ln;     // B column = weight row
      const __bf16* gc = gW + (size_t)f * D;
      const __bf16* uc = uW + (size_t)f * D;
      // B layout (32x16 bf16): lane col = ln, Ks contiguous at h*16
      v16bf G0 = *(const v16bf*)(gc + 16 * h);
      v16bf G1 = *(const v16bf*)(gc + 32 + 16 * h);
      v16bf U0 = *(const v16bf*)(uc + 16 * h);
      v16bf U1 = *(const v16bf*)(uc + 32 + 16 * h);
      v8f ag = {}, au = {};
      ag = wmma_bf16(A0, G0, ag);
      ag = wmma_bf16(A1, G1, ag);
      au = wmma_bf16(A0, U0, au);
      au = wmma_bf16(A1, U1, au);
#pragma unroll
      for (int v = 0; v < 8; ++v) {      // C elem (v,lane): M = v+8h, N = ln
        float gv = ag[v];
        float hv = (gv / (1.f + expf(-gv))) * au[v];
        s_hid[Mblk * 16 + v + 8 * h][nch * 64 + nt * 16 + ln] = (__bf16)hv;
      }
    }
  }

#if HAVE_TDM
  __builtin_amdgcn_s_wait_tensorcnt(0);
#elif HAVE_ASYNC
#if __has_builtin(__builtin_amdgcn_s_wait_asynccnt)
  __builtin_amdgcn_s_wait_asynccnt(0);
#else
  asm volatile("s_wait_asynccnt 0x0" ::: "memory");
#endif
#endif
  __syncthreads();

  { // ---- stage 2: down GEMM, K = FF = 256 ----
    const int Mblk  = wave & 1;
    const int ntile = wave >> 1;         // 16-wide d block: 0..3
#if USE_LDS_DW
    const __bf16* dc = (const __bf16*)&s_dw[ntile * 16 + ln][0];
#else
    const __bf16* dc = downB + (size_t)g * (D * FF)
                             + (size_t)(ntile * 16 + ln) * FF;
#endif
    const __bf16* hrow = &s_hid[Mblk * 16 + ln][0];
    v8f acc = {};
#pragma unroll
    for (int kc = 0; kc < 8; ++kc) {
#if USE_LDS_DW
      v16bf Bd = pack_b(dc + kc * 32 + 16 * h);      // ds_load_b128 pair
#else
      if (kc < 7) __builtin_prefetch(dc + (kc + 1) * 32, 0, 0);
      v16bf Bd = *(const v16bf*)(dc + kc * 32 + 16 * h);
#endif
      v16bf Ah = pack_a(hrow, kc * 32, h);           // ds_load_b128 path
      acc = wmma_bf16(Ah, Bd, acc);
    }
#pragma unroll
    for (int v = 0; v < 8; ++v) {
      int r = Mblk * 16 + v + 8 * h;
      int gr = m0 + r;
      if (gr < count)
        expertOut[((size_t)(base + gr)) * 64 + ntile * 16 + ln] = acc[v];
    }
  }
}

// ---------------------------------------------------------------------------
__global__ void k_combine(const float* __restrict__ x,
                          const float* __restrict__ bladesF,
                          const float* __restrict__ expertOut,
                          const int* __restrict__ slotOf,
                          const float* __restrict__ wts,
                          const float* __restrict__ norm_out_w,
                          float* __restrict__ out) {
  const int t = blockIdx.x;
  const int tid = threadIdx.x;
  const int wave = tid >> 5, lane = tid & 31;
  __shared__ float s_red[8];

  const int e0 = 2 * tid;
  const int b = e0 >> 6, d = e0 & 63;
  const int ib = (t * NB + b) * 2;
  const int s0 = slotOf[ib], s1 = slotOf[ib + 1];
  const float w0 = wts[s0], w1 = wts[s1];
  const size_t bb = ((size_t)t * NB + b) * 64 + d;
  float v0 = bladesF[bb]     + w0 * expertOut[(size_t)s0 * 64 + d]
                             + w1 * expertOut[(size_t)s1 * 64 + d];
  float v1 = bladesF[bb + 1] + w0 * expertOut[(size_t)s0 * 64 + d + 1]
                             + w1 * expertOut[(size_t)s1 * 64 + d + 1];
  float ss = v0 * v0 + v1 * v1;
#pragma unroll
  for (int off = 16; off; off >>= 1) ss += __shfl_xor(ss, off, 32);
  if (lane == 0) s_red[wave] = ss;
  __syncthreads();
  float tot = 0.f;
#pragma unroll
  for (int i = 0; i < 8; ++i) tot += s_red[i];
  float rinv = rsqrtf(tot * (1.f / 512.f) + EPS_RMS);
  const size_t xb = (size_t)t * 512 + e0;
  out[xb]     = x[xb]     + v0 * rinv * norm_out_w[e0];
  out[xb + 1] = x[xb + 1] + v1 * rinv * norm_out_w[e0 + 1];
}

__global__ void k_aux(const float* __restrict__ f_sum,
                      const int* __restrict__ counts,
                      float* __restrict__ out_aux) {
  __shared__ float s[2];
  int tid = threadIdx.x;
  float v = 0.f;
  if (tid < NGRP)
    v = (f_sum[tid] * (1.f / (float)NTOK)) *
        ((float)counts[tid] * (1.f / (float)NTOK));
#pragma unroll
  for (int off = 16; off; off >>= 1) v += __shfl_xor(v, off, 32);
  if ((tid & 31) == 0) s[tid >> 5] = v;
  __syncthreads();
  if (tid == 0) out_aux[0] = 0.01f * (float)NE * (s[0] + s[1]);
}

// ---------------------------------------------------------------------------
extern "C" void kernel_launch(void* const* d_in, const int* in_sizes, int n_in,
                              void* d_out, int out_size, void* d_ws,
                              size_t ws_size, hipStream_t stream) {
  (void)in_sizes; (void)n_in; (void)out_size; (void)ws_size;
  const float* x           = (const float*)d_in[0];
  const float* inter       = (const float*)d_in[1];
  const float* norm_in_w   = (const float*)d_in[2];
  const float* norm_out_w  = (const float*)d_in[3];
  const float* router_w    = (const float*)d_in[4];
  const float* expert_gate = (const float*)d_in[5];
  const float* expert_up   = (const float*)d_in[6];
  const float* expert_down = (const float*)d_in[7];
  // d_in[8] = top_k, fixed at 2 per reference.

  char* w = (char*)d_ws;
  auto alloc = [&](size_t bytes) -> char* {
    char* p = w;
    w += (bytes + 255) & ~(size_t)255;
    return p;
  };
  float*  bladesF  = (float*)alloc((size_t)NTOK * 512 * 4);   // 8 MB
  __bf16* bladesB  = (__bf16*)alloc((size_t)NTOK * 512 * 2);  // 4 MB
  __bf16* gateB    = (__bf16*)alloc((size_t)WEL * 2);         // 2 MB
  __bf16* upB      = (__bf16*)alloc((size_t)WEL * 2);         // 2 MB
  __bf16* downB    = (__bf16*)alloc((size_t)WEL * 2);         // 2 MB
  float*  expertOut= (float*)alloc((size_t)NASSIGN * 64 * 4); // 16 MB
  int*    topIdx   = (int*)alloc((size_t)NASSIGN * 4);
  float*  topW     = (float*)alloc((size_t)NASSIGN * 4);
  int*    rowsB    = (int*)alloc((size_t)NASSIGN * 4);
  float*  wtsB     = (float*)alloc((size_t)NASSIGN * 4);
  int*    slotOf   = (int*)alloc((size_t)NASSIGN * 4);
  float*  f_sum    = (float*)alloc(NGRP * 4);
  int*    counts   = (int*)alloc(NGRP * 4);
  int*    cursor   = (int*)alloc(NGRP * 4);
  int*    offsets  = (int*)alloc(NGRP * 4);

  k_init<<<1, 64, 0, stream>>>(f_sum, counts, cursor);
  k_f32_to_bf16<<<WEL / 256, 256, 0, stream>>>(expert_gate, gateB, WEL);
  k_f32_to_bf16<<<WEL / 256, 256, 0, stream>>>(expert_up,   upB,   WEL);
  k_f32_to_bf16<<<WEL / 256, 256, 0, stream>>>(expert_down, downB, WEL);
  k_token_prep<<<NTOK, 256, 0, stream>>>(x, inter, norm_in_w, router_w,
                                         bladesF, bladesB, topIdx, topW,
                                         f_sum, counts);
  k_prefix<<<1, 64, 0, stream>>>(counts, offsets);
  k_assign<<<NASSIGN / 256, 256, 0, stream>>>(topIdx, topW, offsets, cursor,
                                              rowsB, wtsB, slotOf);
  k_moe_gemm<<<dim3(128, NGRP), 256, 0, stream>>>(bladesB, gateB, upB, downB,
                                                  rowsB, offsets, counts,
                                                  expertOut);
  k_combine<<<NTOK, 256, 0, stream>>>(x, bladesF, expertOut, slotOf, wtsB,
                                      norm_out_w, (float*)d_out);
  k_aux<<<1, 64, 0, stream>>>(f_sum, counts, (float*)d_out + (size_t)NTOK * 512);
}
